// ConditionalVectorQuantization_13322988552584
// MI455X (gfx1250) — matrix operand order repaired and checked by reading, not compiled, not run on hardware
//
#include <hip/hip_runtime.h>

typedef __attribute__((ext_vector_type(16))) __bf16 v16bf;
typedef __attribute__((ext_vector_type(8)))  float  v8f;

#define G        4
#define DIM      128
#define CBSZ     512
#define ROWS_PB  128      // rows of n per block
#define THREADS  256      // 8 waves
#define LDS_CB_BYTES   (CBSZ * DIM * 2)        // 131072 : bf16 codebook, B-frag order
#define LDS_CB2_BYTES  (CBSZ * 4)              // 2048   : fp32 |cb|^2
#define LDS_IDX_BYTES  (8 * 16 * 4)            // 512    : per-wave argmin indices
#define LDS_TOTAL      (LDS_CB_BYTES + LDS_CB2_BYTES + LDS_IDX_BYTES)

__device__ __forceinline__ unsigned pk_bf2(float a, float b) {
  unsigned short ua = __builtin_bit_cast(unsigned short, (__bf16)a);
  unsigned short ub = __builtin_bit_cast(unsigned short, (__bf16)b);
  return (unsigned)ua | ((unsigned)ub << 16);
}

__global__ __launch_bounds__(THREADS)
void ConditionalVectorQuantization_13322988552584_kernel(
    const float* __restrict__ x,       // (n, 4, 128)
    const float* __restrict__ cbk,     // (4, 512, 128)
    float* __restrict__ out,           // xhat | one_hot | index
    int n) {
  extern __shared__ char smem[];
  __bf16* cbf  = (__bf16*)smem;                            // 512*128 bf16, fragment order
  float*  cb2s = (float*)(smem + LDS_CB_BYTES);            // 512 floats
  int*    idxs = (int*)(smem + LDS_CB_BYTES + LDS_CB2_BYTES); // 8*16 ints

  const int g    = blockIdx.y;
  const int tid  = threadIdx.x;
  const int lane = tid & 31;
  const int w    = tid >> 5;
  const int r16  = lane & 15;
  const int half = lane >> 4;

  // ---------------- stage codebook: fp32 -> bf16 B-fragments + exact cb2 ----------------
  // B-frag for tile (c = code/16, t = d/32): lane = (code&15) + 16*((d>>3)&1),
  // packed element = (d&7) + 8*((d>>4)&1).  LDS element offset = ((c*4+t)*32 + lane)*16 + element.
  for (int kk = 0; kk < 2; ++kk) {
    int k = tid * 2 + kk;                       // 512 codes / 256 threads
    const float* src = cbk + ((size_t)g * CBSZ + k) * DIM;
    float ss = 0.f;
    int c = k >> 4, lc = k & 15;
    #pragma unroll
    for (int t = 0; t < 4; ++t) {
      #pragma unroll
      for (int q = 0; q < 8; ++q) {             // 4 floats per step, d = t*32 + q*4
        float4 v = ((const float4*)src)[t * 8 + q];
        ss += v.x * v.x + v.y * v.y + v.z * v.z + v.w * v.w;
        int ln   = lc + (((q >> 1) & 1) << 4);
        int sub  = ((q & 1) << 2) | ((q >> 2) << 3);
        int base = (((c * 4 + t) * 32) + ln) * 16 + sub;
        *(unsigned*)(cbf + base)     = pk_bf2(v.x, v.y);
        *(unsigned*)(cbf + base + 2) = pk_bf2(v.z, v.w);
      }
    }
    cb2s[k] = ss;
  }
  __syncthreads();

  // ---------------- load 16 rows of x into A-fragments + exact x2 ----------------
  // A-frag 16x32 bf16: lanes 0-15 row M=lane, K = {0..7,16..23}; lanes 16-31 same rows,
  // K = {8..15,24..31}. Per lane: two contiguous 8-float runs.
  const int rowBase = blockIdx.x * ROWS_PB + w * 16;
  const float* xr = x + (((size_t)(rowBase + r16)) * G + g) * DIM;
  v16bf afrag[4];
  float xs = 0.f;
  #pragma unroll
  for (int t = 0; t < 4; ++t) {
    int d0 = t * 32 + half * 8;
    float4 a0 = *(const float4*)(xr + d0);
    float4 a1 = *(const float4*)(xr + d0 + 4);
    float4 a2 = *(const float4*)(xr + d0 + 16);
    float4 a3 = *(const float4*)(xr + d0 + 20);
    xs += a0.x*a0.x + a0.y*a0.y + a0.z*a0.z + a0.w*a0.w
        + a1.x*a1.x + a1.y*a1.y + a1.z*a1.z + a1.w*a1.w
        + a2.x*a2.x + a2.y*a2.y + a2.z*a2.z + a2.w*a2.w
        + a3.x*a3.x + a3.y*a3.y + a3.z*a3.z + a3.w*a3.w;
    afrag[t][0]  = (__bf16)a0.x; afrag[t][1]  = (__bf16)a0.y;
    afrag[t][2]  = (__bf16)a0.z; afrag[t][3]  = (__bf16)a0.w;
    afrag[t][4]  = (__bf16)a1.x; afrag[t][5]  = (__bf16)a1.y;
    afrag[t][6]  = (__bf16)a1.z; afrag[t][7]  = (__bf16)a1.w;
    afrag[t][8]  = (__bf16)a2.x; afrag[t][9]  = (__bf16)a2.y;
    afrag[t][10] = (__bf16)a2.z; afrag[t][11] = (__bf16)a2.w;
    afrag[t][12] = (__bf16)a3.x; afrag[t][13] = (__bf16)a3.y;
    afrag[t][14] = (__bf16)a3.z; afrag[t][15] = (__bf16)a3.w;
  }
  float x2row = xs + __shfl_xor(xs, 16, 32);   // full-row |x|^2, replicated in both halves
  // C/D layout: lane holds rows v + 8*half, col = lane&15 -> fetch x2 for my 8 row slots
  float x2v[8];
  #pragma unroll
  for (int v = 0; v < 8; ++v) x2v[v] = __shfl(x2row, half * 8 + v, 32);

  // ---------------- GEMM over 32 code tiles + running argmin ----------------
  float minv[8];
  int   mini[8];
  #pragma unroll
  for (int v = 0; v < 8; ++v) { minv[v] = 3.402823466e38f; mini[v] = 0; }

  for (int c = 0; c < 32; ++c) {
    v8f acc = {0.f, 0.f, 0.f, 0.f, 0.f, 0.f, 0.f, 0.f};
    const __bf16* bp = cbf + ((size_t)(c * 4) * 32 + lane) * 16;
    #pragma unroll
    for (int t = 0; t < 4; ++t) {
      v16bf bfrag = *(const v16bf*)(bp + t * 512);
      acc = __builtin_amdgcn_wmma_f32_16x16x32_bf16(
          false, afrag[t], false, bfrag, (short)0, acc, false, false);
    }
    float cb2c = cb2s[c * 16 + r16];
    int   col  = c * 16 + r16;
    #pragma unroll
    for (int v = 0; v < 8; ++v) {
      float d = x2v[v] + cb2c - 2.0f * acc[v];
      if (d < minv[v]) { minv[v] = d; mini[v] = col; }   // idx increasing: keeps first
    }
  }

  // reduce across the 16 columns (lexicographic (val,idx) -> first-index argmin)
  #pragma unroll
  for (int m = 1; m <= 8; m <<= 1) {
    #pragma unroll
    for (int v = 0; v < 8; ++v) {
      float ov = __shfl_xor(minv[v], m, 32);
      int   oi = __shfl_xor(mini[v], m, 32);
      if (ov < minv[v] || (ov == minv[v] && oi < mini[v])) { minv[v] = ov; mini[v] = oi; }
    }
  }
  int* widx = idxs + w * 16;
  if (r16 == 0) {
    #pragma unroll
    for (int v = 0; v < 8; ++v) widx[half * 8 + v] = mini[v];  // lane0: rows0-7, lane16: rows8-15
  }

  // ---------------- outputs ----------------
  float* out_xhat = out;
  float* out_oh   = out + (size_t)n * G * DIM;
  float* out_idx  = out_oh + (size_t)n * G * CBSZ;

  if (lane < 16) out_idx[(size_t)(rowBase + lane) * G + g] = (float)widx[lane];

  for (int r = 0; r < 16; ++r) {
    int rowN = rowBase + r;
    int id   = widx[r];
    // x_hat: exact fp32 codebook row gather (128 floats = 32 lanes x float4)
    float4 cv = *(const float4*)(cbk + ((size_t)g * CBSZ + id) * DIM + lane * 4);
    *(float4*)(out_xhat + ((size_t)rowN * G + g) * DIM + lane * 4) = cv;
    // one_hot: full 512-float row, single pass
    float* ohp = out_oh + ((size_t)rowN * G + g) * CBSZ;
    #pragma unroll
    for (int q = 0; q < 4; ++q) {
      int pos = q * 128 + lane * 4;
      float4 z = make_float4(0.f, 0.f, 0.f, 0.f);
      int dsel = id - pos;
      if (dsel >= 0 && dsel < 4) ((float*)&z)[dsel] = 1.0f;
      *(float4*)(ohp + pos) = z;
    }
  }
}

extern "C" void kernel_launch(void* const* d_in, const int* in_sizes, int n_in,
                              void* d_out, int out_size, void* d_ws, size_t ws_size,
                              hipStream_t stream) {
  const float* x   = (const float*)d_in[0];
  const float* cbk = (const float*)d_in[1];
  float* out = (float*)d_out;
  const int n = in_sizes[0] / (G * DIM);   // 32768

  // dynamic LDS > 64KB needs an opt-in (not a stream op; capture-safe)
  static bool attr_set = false;
  if (!attr_set) {
    hipFuncSetAttribute(
        (const void*)ConditionalVectorQuantization_13322988552584_kernel,
        hipFuncAttributeMaxDynamicSharedMemorySize, LDS_TOTAL);
    attr_set = true;
  }

  dim3 grid(n / ROWS_PB, G);
  ConditionalVectorQuantization_13322988552584_kernel
      <<<grid, THREADS, LDS_TOTAL, stream>>>(x, cbk, out, n);
}
